// TransformerEncoder_39848706572291
// MI455X (gfx1250) — matrix operand order repaired and checked by reading, not compiled, not run on hardware
//
#include <hip/hip_runtime.h>
#include <hip/hip_bf16.h>
#include <math.h>

// Problem constants (fixed by the reference)
static constexpr int SEQ = 2048;
static constexpr int DIM = 1024;
static constexpr int NH  = 8;
static constexpr int HD  = 128;   // head dim

typedef __attribute__((ext_vector_type(16))) __bf16 v16bf;
typedef __attribute__((ext_vector_type(8)))  __bf16 v8bf;
typedef __attribute__((ext_vector_type(8)))  float  v8f;
typedef int v4i_vec __attribute__((vector_size(16)));   // matches builtin param type

#ifndef __has_builtin
#define __has_builtin(x) 0
#endif

#if defined(__HIP_DEVICE_COMPILE__) && __has_builtin(__builtin_amdgcn_global_load_async_to_lds_b128)
#define USE_ASYNC_LDS 1
#else
#define USE_ASYNC_LDS 0
#endif

#define AS1 __attribute__((address_space(1)))
#define AS3 __attribute__((address_space(3)))

#if USE_ASYNC_LDS
static __device__ __forceinline__ void async_copy16(const __bf16* g, __bf16* l) {
  __builtin_amdgcn_global_load_async_to_lds_b128(
      (AS1 v4i_vec*)g, (AS3 v4i_vec*)l, 0, 0);
}
static __device__ __forceinline__ void wait_async_all() {
#if __has_builtin(__builtin_amdgcn_s_wait_asynccnt)
  __builtin_amdgcn_s_wait_asynccnt(0);
#else
  asm volatile("s_wait_asynccnt 0x0" ::: "memory");
#endif
}
#endif

// Build a 16x32 A-fragment / 32x16 B-fragment register image from two
// 16-byte chunks (keeps every LDS/global access 16B-aligned).
static __device__ __forceinline__ v16bf ld16pair(const __bf16* p0, const __bf16* p1) {
  v8bf lo = *(const v8bf*)p0;
  v8bf hi = *(const v8bf*)p1;
  v16bf r;
#pragma unroll
  for (int e = 0; e < 8; ++e) { r[e] = lo[e]; r[8 + e] = hi[e]; }
  return r;
}

// ---------------------------------------------------------------------------
// Generic bf16 WMMA GEMM: C[M,N] = A[M,K] @ Bt^T (+bias), batched over z.
// Bt is the PRE-TRANSPOSED weight: Bt[n][k], row stride ldbt.
// Block tile 128x128, BK=32; 8 waves, each wave computes 64x32 (4x2 tiles).
// Double-buffered LDS, async global->LDS when available.
// ---------------------------------------------------------------------------
static constexpr int BM = 128;
static constexpr int BN = 128;
static constexpr int BK = 32;
static constexpr int LSTR = 40;  // LDS row stride (elems): 80B, 16B-aligned, padded

__global__ __launch_bounds__(256)
void gemm_bf16_wmma(const __bf16* __restrict__ A, long long strideA,
                    const __bf16* __restrict__ Bt, long long strideB,
                    const float* __restrict__ bias, long long strideBias,
                    float* __restrict__ outF, long long strideOF,
                    __bf16* __restrict__ outB, long long strideOB,
                    __bf16* __restrict__ outBT, long long strideOBT,
                    int M, int N, int K, int lda, int ldbt, int ldc, int ldct) {
  __shared__ __bf16 As[2][BM][LSTR];
  __shared__ __bf16 Bs[2][BN][LSTR];  // Bs[n][k] (already transposed in global)

  const int z = blockIdx.z;
  A  += (size_t)z * strideA;
  Bt += (size_t)z * strideB;
  const float* biasz = bias ? bias + (size_t)z * strideBias : nullptr;
  float*  outFz  = outF  ? outF  + (size_t)z * strideOF  : nullptr;
  __bf16* outBz  = outB  ? outB  + (size_t)z * strideOB  : nullptr;
  __bf16* outBTz = outBT ? outBT + (size_t)z * strideOBT : nullptr;

  const int m0 = blockIdx.y * BM;
  const int n0 = blockIdx.x * BN;
  const int tid  = threadIdx.x;
  const int wave = tid >> 5;
  const int lane = tid & 31;
  const int wm = wave >> 2;        // 0..1
  const int wn = wave & 3;         // 0..3
  const int half = lane >> 4;      // 0..1
  const int lr   = lane & 15;      // 0..15
  const int ka0  = half ? 8 : 0;   // A-frag K sub-chunk base per ISA layout

  v8f acc[4][2];
#pragma unroll
  for (int mt = 0; mt < 4; ++mt)
#pragma unroll
    for (int nt = 0; nt < 2; ++nt)
#pragma unroll
      for (int e = 0; e < 8; ++e) acc[mt][nt][e] = 0.0f;

#if USE_ASYNC_LDS
  auto stage_async = [&](int buf_, int k0_) {
#pragma unroll
    for (int c = 0; c < 2; ++c) {
      int idx = tid + c * 256;       // 0..511
      int r   = idx >> 2;            // 0..127
      int kc  = (idx & 3) * 8;       // 0,8,16,24
      async_copy16(A  + (size_t)(m0 + r) * lda  + k0_ + kc, &As[buf_][r][kc]);
      async_copy16(Bt + (size_t)(n0 + r) * ldbt + k0_ + kc, &Bs[buf_][r][kc]);
    }
  };
#else
  uint4 ra[2], rb[2];
  auto load_regs = [&](int k0_) {
#pragma unroll
    for (int c = 0; c < 2; ++c) {
      int idx = tid + c * 256;
      int r   = idx >> 2;
      int kc  = (idx & 3) * 8;
      ra[c] = *(const uint4*)(A  + (size_t)(m0 + r) * lda  + k0_ + kc);
      rb[c] = *(const uint4*)(Bt + (size_t)(n0 + r) * ldbt + k0_ + kc);
    }
  };
  auto store_regs = [&](int buf_) {
#pragma unroll
    for (int c = 0; c < 2; ++c) {
      int idx = tid + c * 256;
      int r   = idx >> 2;
      int kc  = (idx & 3) * 8;
      *(uint4*)&As[buf_][r][kc] = ra[c];
      *(uint4*)&Bs[buf_][r][kc] = rb[c];
    }
  };
#endif

  int buf = 0;
#if USE_ASYNC_LDS
  stage_async(0, 0);
#else
  load_regs(0);
  store_regs(0);
#endif

  for (int k0 = 0; k0 < K; k0 += BK) {
    const bool more = (k0 + BK) < K;
#if USE_ASYNC_LDS
    wait_async_all();
#endif
    __syncthreads();
#if USE_ASYNC_LDS
    if (more) stage_async(buf ^ 1, k0 + BK);   // overlap with compute below
#else
    if (more) load_regs(k0 + BK);              // issue loads; store after compute
#endif

    // --- fragments + WMMAs from LDS[buf] ---
    v16bf af[4];
#pragma unroll
    for (int mt = 0; mt < 4; ++mt) {
      int row = wm * 64 + mt * 16 + lr;
      af[mt] = ld16pair(&As[buf][row][ka0], &As[buf][row][ka0 + 16]);
    }
    v16bf bfr[2];
#pragma unroll
    for (int nt = 0; nt < 2; ++nt) {
      int col = wn * 32 + nt * 16 + lr;
      bfr[nt] = ld16pair(&Bs[buf][col][ka0], &Bs[buf][col][ka0 + 16]);
    }
#pragma unroll
    for (int mt = 0; mt < 4; ++mt)
#pragma unroll
      for (int nt = 0; nt < 2; ++nt)
        acc[mt][nt] = __builtin_amdgcn_wmma_f32_16x16x32_bf16(
            false, af[mt], false, bfr[nt], (short)0, acc[mt][nt], false, false);

#if !USE_ASYNC_LDS
    if (more) store_regs(buf ^ 1);
#endif
    buf ^= 1;
  }

  // --- epilogue: C tile layout (VGPR r: lanes0-15 M=r, lanes16-31 M=8+r) ---
#pragma unroll
  for (int mt = 0; mt < 4; ++mt) {
#pragma unroll
    for (int nt = 0; nt < 2; ++nt) {
      int rbase = m0 + wm * 64 + mt * 16 + half * 8;
      int col   = n0 + wn * 32 + nt * 16 + lr;
      float bb = biasz ? biasz[col] : 0.0f;
#pragma unroll
      for (int r = 0; r < 8; ++r) {
        float v = acc[mt][nt][r] + bb;
        if (outFz)  outFz[(size_t)(rbase + r) * ldc + col] = v;
        if (outBz)  outBz[(size_t)(rbase + r) * ldc + col] = (__bf16)v;
        if (outBTz) outBTz[(size_t)col * ldct + (rbase + r)] = (__bf16)v;
      }
    }
  }
}

// ---------------------------------------------------------------------------
// Flash attention (online softmax), one head per blockIdx.y.
// Block = 128 threads = 4 waves; each wave owns a 16-query tile.
// K block and transposed-V block staged in double-buffered LDS (async path).
// Writes head-concatenated output: Out[q, h*HD + dk], bf16.
// ---------------------------------------------------------------------------
__global__ __launch_bounds__(128)
void flash_attn_bf16(const __bf16* __restrict__ Q,
                     const __bf16* __restrict__ Km,
                     const __bf16* __restrict__ Vt_g,   // [H][HD][SEQ]
                     __bf16* __restrict__ Out, float scale) {
  constexpr int KSTR = HD + 8;  // 136 elems = 272B rows: 16B aligned, bank-spread
  constexpr int VSTR = 40;      // 80B rows
  __shared__ __bf16 Ks[2][32][KSTR];   // keys x head-dim
  __shared__ __bf16 Vs[2][HD][VSTR];   // head-dim x keys (transposed V)
  __shared__ __bf16 Ps[4][16][32];     // per-wave probability patch

  const int h    = blockIdx.y;
  const int tid  = threadIdx.x;
  const int wave = tid >> 5;
  const int lane = tid & 31;
  const int half = lane >> 4;
  const int lr   = lane & 15;
  const int ka0  = half ? 8 : 0;
  const int qbase = blockIdx.x * 64 + wave * 16;

  const __bf16* Qh  = Q    + (size_t)h * SEQ * HD;
  const __bf16* Kh  = Km   + (size_t)h * SEQ * HD;
  const __bf16* Vth = Vt_g + (size_t)h * HD * SEQ;

  // Q fragments: 4 chunks of K=32 along head dim (HD=128)
  v16bf qf[4];
  {
    const __bf16* qrow = Qh + (size_t)(qbase + lr) * HD;
#pragma unroll
    for (int c = 0; c < 4; ++c)
      qf[c] = ld16pair(qrow + c * 32 + ka0, qrow + c * 32 + ka0 + 16);
  }

  v8f o[8];
#pragma unroll
  for (int d = 0; d < 8; ++d)
#pragma unroll
    for (int e = 0; e < 8; ++e) o[d][e] = 0.0f;
  float mrow[8], lsum[8];
#pragma unroll
  for (int r = 0; r < 8; ++r) { mrow[r] = -INFINITY; lsum[r] = 0.0f; }

#if USE_ASYNC_LDS
  auto stage_async = [&](int buf_, int kb_) {
#pragma unroll
    for (int c = 0; c < 4; ++c) {
      int idx = tid + c * 128;          // 0..511
      int key = idx >> 4;               // 0..31
      int dkc = (idx & 15) * 8;         // 0..120
      async_copy16(Kh + (size_t)(kb_ + key) * HD + dkc, &Ks[buf_][key][dkc]);
      int dk = idx >> 2;                // 0..127
      int kc = (idx & 3) * 8;           // 0..24
      async_copy16(Vth + (size_t)dk * SEQ + kb_ + kc, &Vs[buf_][dk][kc]);
    }
  };
#else
  uint4 rk[4], rv[4];
  auto load_regs = [&](int kb_) {
#pragma unroll
    for (int c = 0; c < 4; ++c) {
      int idx = tid + c * 128;
      rk[c] = *(const uint4*)(Kh + (size_t)(kb_ + (idx >> 4)) * HD + (idx & 15) * 8);
      rv[c] = *(const uint4*)(Vth + (size_t)(idx >> 2) * SEQ + kb_ + (idx & 3) * 8);
    }
  };
  auto store_regs = [&](int buf_) {
#pragma unroll
    for (int c = 0; c < 4; ++c) {
      int idx = tid + c * 128;
      *(uint4*)&Ks[buf_][idx >> 4][(idx & 15) * 8] = rk[c];
      *(uint4*)&Vs[buf_][idx >> 2][(idx & 3) * 8]  = rv[c];
    }
  };
#endif

  int buf = 0;
#if USE_ASYNC_LDS
  stage_async(0, 0);
#else
  load_regs(0);
  store_regs(0);
#endif

  for (int kb = 0; kb < SEQ; kb += 32) {
    const bool more = (kb + 32) < SEQ;
#if USE_ASYNC_LDS
    wait_async_all();
#endif
    __syncthreads();
#if USE_ASYNC_LDS
    if (more) stage_async(buf ^ 1, kb + 32);
#else
    if (more) load_regs(kb + 32);
#endif

    // scores S = Q K^T for two 16-key sub-tiles (K rows from LDS)
    v8f s0, s1;
#pragma unroll
    for (int e = 0; e < 8; ++e) { s0[e] = 0.0f; s1[e] = 0.0f; }
#pragma unroll
    for (int c = 0; c < 4; ++c) {
      const __bf16* k0p = &Ks[buf][lr][c * 32 + ka0];
      const __bf16* k1p = &Ks[buf][16 + lr][c * 32 + ka0];
      v16bf kf0 = ld16pair(k0p, k0p + 16);
      v16bf kf1 = ld16pair(k1p, k1p + 16);
      s0 = __builtin_amdgcn_wmma_f32_16x16x32_bf16(false, qf[c], false, kf0, (short)0, s0, false, false);
      s1 = __builtin_amdgcn_wmma_f32_16x16x32_bf16(false, qf[c], false, kf1, (short)0, s1, false, false);
    }

    // online softmax per query row (row r lives in 16 lanes of this half)
#pragma unroll
    for (int r = 0; r < 8; ++r) {
      float a0 = s0[r] * scale;
      float a1 = s1[r] * scale;
      float mx = fmaxf(a0, a1);
#pragma unroll
      for (int m = 1; m < 16; m <<= 1) mx = fmaxf(mx, __shfl_xor(mx, m, 32));
      float mnew = fmaxf(mrow[r], mx);
      float p0 = __expf(a0 - mnew);
      float p1 = __expf(a1 - mnew);
      float ps = p0 + p1;
#pragma unroll
      for (int m = 1; m < 16; m <<= 1) ps += __shfl_xor(ps, m, 32);
      float alpha = __expf(mrow[r] - mnew);
      lsum[r] = lsum[r] * alpha + ps;
      mrow[r] = mnew;
#pragma unroll
      for (int d = 0; d < 8; ++d) o[d][r] *= alpha;
      int rl = half ? 8 + r : r;
      Ps[wave][rl][lr]      = (__bf16)p0;
      Ps[wave][rl][16 + lr] = (__bf16)p1;
    }

    // re-layout P (C-layout -> A-fragment) via per-wave LDS patch (in-order DS)
    v16bf pf = ld16pair(&Ps[wave][lr][ka0], &Ps[wave][lr][ka0 + 16]);

    // O += P @ V  (8 dk tiles of 16; V columns are contiguous in Vs)
#pragma unroll
    for (int d = 0; d < 8; ++d) {
      const __bf16* vp = &Vs[buf][d * 16 + lr][ka0];
      v16bf vf = ld16pair(vp, vp + 16);
      o[d] = __builtin_amdgcn_wmma_f32_16x16x32_bf16(false, pf, false, vf, (short)0, o[d], false, false);
    }

#if !USE_ASYNC_LDS
    if (more) store_regs(buf ^ 1);
#endif
    buf ^= 1;
  }

  // normalize and write concatenated heads: Out[q, h*HD + dk]
#pragma unroll
  for (int r = 0; r < 8; ++r) {
    float inv = 1.0f / lsum[r];
    int q = qbase + (half ? 8 + r : r);
#pragma unroll
    for (int d = 0; d < 8; ++d)
      Out[(size_t)q * (NH * HD) + h * HD + d * 16 + lr] = (__bf16)(o[d][r] * inv);
  }
}

// ---------------------------------------------------------------------------
// Fused residual + LayerNorm: out = LN(x + y) * gamma + beta (f32 + bf16 out)
// ---------------------------------------------------------------------------
__global__ __launch_bounds__(256)
void ln_residual(const float* __restrict__ X, const float* __restrict__ Y,
                 const float* __restrict__ gamma, const float* __restrict__ beta,
                 float* __restrict__ outF, __bf16* __restrict__ outB) {
  __shared__ float red[256];
  const int row = blockIdx.x;
  const int tid = threadIdx.x;
  const float* xr = X + (size_t)row * DIM;
  const float* yr = Y + (size_t)row * DIM;

  float v[4];
  float s = 0.0f;
#pragma unroll
  for (int i = 0; i < 4; ++i) { v[i] = xr[tid + i * 256] + yr[tid + i * 256]; s += v[i]; }
  red[tid] = s; __syncthreads();
#pragma unroll
  for (int st = 128; st > 0; st >>= 1) { if (tid < st) red[tid] += red[tid + st]; __syncthreads(); }
  float mean = red[0] / (float)DIM; __syncthreads();

  float sq = 0.0f;
#pragma unroll
  for (int i = 0; i < 4; ++i) { float d = v[i] - mean; sq += d * d; }
  red[tid] = sq; __syncthreads();
#pragma unroll
  for (int st = 128; st > 0; st >>= 1) { if (tid < st) red[tid] += red[tid + st]; __syncthreads(); }
  float rstd = rsqrtf(red[0] / (float)DIM + 1e-5f);

#pragma unroll
  for (int i = 0; i < 4; ++i) {
    int c = tid + i * 256;
    float oo = (v[i] - mean) * rstd * gamma[c] + beta[c];
    outF[(size_t)row * DIM + c] = oo;
    outB[(size_t)row * DIM + c] = (__bf16)oo;
  }
}

// ---------------------------------------------------------------------------
// Elementwise / layout helpers
// ---------------------------------------------------------------------------
__global__ void cvt_bf16(const float* __restrict__ in, __bf16* __restrict__ outb,
                         float* __restrict__ outf, int n) {
  for (int i = blockIdx.x * blockDim.x + threadIdx.x; i < n; i += gridDim.x * blockDim.x) {
    float v = in[i];
    outb[i] = (__bf16)v;
    if (outf) outf[i] = v;
  }
}

// out[c][r] = (bf16) in[r][c], batched over z; R,C multiples of 32
__global__ __launch_bounds__(256)
void transpose_cvt_bf16(const float* __restrict__ in, __bf16* __restrict__ out,
                        int R, int C) {
  __shared__ float t[32][33];
  const int b = blockIdx.z;
  const float* inb = in + (size_t)b * R * C;
  __bf16* outb = out + (size_t)b * R * C;
  const int c0 = blockIdx.x * 32, r0 = blockIdx.y * 32;
  const int tx = threadIdx.x, ty = threadIdx.y;  // 32 x 8
#pragma unroll
  for (int i = 0; i < 4; ++i)
    t[ty + i * 8][tx] = inb[(size_t)(r0 + ty + i * 8) * C + c0 + tx];
  __syncthreads();
#pragma unroll
  for (int i = 0; i < 4; ++i)
    outb[(size_t)(c0 + ty + i * 8) * R + r0 + tx] = (__bf16)t[tx][ty + i * 8];
}

__global__ void copy_f32(const float* __restrict__ in, float* __restrict__ out, int n) {
  for (int i = blockIdx.x * blockDim.x + threadIdx.x; i < n; i += gridDim.x * blockDim.x)
    out[i] = in[i];
}

// ---------------------------------------------------------------------------
// Host orchestration
// ---------------------------------------------------------------------------
extern "C" void kernel_launch(void* const* d_in, const int* in_sizes, int n_in,
                              void* d_out, int out_size, void* d_ws, size_t ws_size,
                              hipStream_t stream) {
  (void)in_sizes; (void)n_in; (void)out_size; (void)ws_size;
  const float* token = (const float*)d_in[0];
  const float* Wq    = (const float*)d_in[1];
  const float* bq    = (const float*)d_in[2];
  const float* Wk    = (const float*)d_in[3];
  const float* bk    = (const float*)d_in[4];
  const float* Wv    = (const float*)d_in[5];
  const float* bv    = (const float*)d_in[6];
  const float* Wo    = (const float*)d_in[7];
  const float* bo    = (const float*)d_in[8];
  const float* Wf    = (const float*)d_in[9];
  const float* bff   = (const float*)d_in[10];
  const float* gamma = (const float*)d_in[11];
  const float* beta  = (const float*)d_in[12];
  float* out = (float*)d_out;

  char* wsp = (char*)d_ws;
  auto alloc = [&](size_t bytes) -> void* {
    void* p = (void*)wsp;
    wsp += (bytes + 255) & ~(size_t)255;
    return p;
  };
  // Pre-transposed bf16 weights: Wqt/Wkt/Wvt [NH][HD][DIM], Wot [DIM][NH*HD], Wft [DIM][DIM]
  __bf16* Wqt = (__bf16*)alloc((size_t)NH * DIM * HD * 2);
  __bf16* Wkt = (__bf16*)alloc((size_t)NH * DIM * HD * 2);
  __bf16* Wvt = (__bf16*)alloc((size_t)NH * DIM * HD * 2);
  __bf16* Wot = (__bf16*)alloc((size_t)NH * HD * DIM * 2);
  __bf16* Wft = (__bf16*)alloc((size_t)DIM * DIM * 2);
  __bf16* xb  = (__bf16*)alloc((size_t)SEQ * DIM * 2);
  __bf16* Qb  = (__bf16*)alloc((size_t)NH * SEQ * HD * 2);
  __bf16* Kb  = (__bf16*)alloc((size_t)NH * SEQ * HD * 2);
  __bf16* Vtg = (__bf16*)alloc((size_t)NH * HD * SEQ * 2);  // transposed V
  __bf16* cat = (__bf16*)alloc((size_t)SEQ * DIM * 2);
  __bf16* ab  = (__bf16*)alloc((size_t)SEQ * DIM * 2);
  float* x_f      = (float*)alloc((size_t)SEQ * DIM * 4);
  float* attn_out = (float*)alloc((size_t)SEQ * DIM * 4);
  float* a_f      = (float*)alloc((size_t)SEQ * DIM * 4);
  float* ff       = (float*)alloc((size_t)SEQ * DIM * 4);

  // one-time weight transpose+convert (weights shared across all 6 layers)
  transpose_cvt_bf16<<<dim3(HD / 32, DIM / 32, NH), dim3(32, 8), 0, stream>>>(Wq, Wqt, DIM, HD);
  transpose_cvt_bf16<<<dim3(HD / 32, DIM / 32, NH), dim3(32, 8), 0, stream>>>(Wk, Wkt, DIM, HD);
  transpose_cvt_bf16<<<dim3(HD / 32, DIM / 32, NH), dim3(32, 8), 0, stream>>>(Wv, Wvt, DIM, HD);
  transpose_cvt_bf16<<<dim3(DIM / 32, (NH * HD) / 32, 1), dim3(32, 8), 0, stream>>>(Wo, Wot, NH * HD, DIM);
  transpose_cvt_bf16<<<dim3(DIM / 32, DIM / 32, 1), dim3(32, 8), 0, stream>>>(Wf, Wft, DIM, DIM);
  cvt_bf16<<<2048, 256, 0, stream>>>(token, xb, x_f, SEQ * DIM);

  const float scale = 0.08838834764831845f;  // 1/sqrt(128)

  for (int layer = 0; layer < 6; ++layer) {
    dim3 gqkv(HD / BN, SEQ / BM, NH);
    // Q, K: row-major bf16 outputs
    gemm_bf16_wmma<<<gqkv, 256, 0, stream>>>(
        xb, 0LL, Wqt, (long long)HD * DIM, bq, (long long)HD,
        nullptr, 0LL, Qb, (long long)SEQ * HD, nullptr, 0LL,
        SEQ, HD, DIM, DIM, DIM, HD, 0);
    gemm_bf16_wmma<<<gqkv, 256, 0, stream>>>(
        xb, 0LL, Wkt, (long long)HD * DIM, bk, (long long)HD,
        nullptr, 0LL, Kb, (long long)SEQ * HD, nullptr, 0LL,
        SEQ, HD, DIM, DIM, DIM, HD, 0);
    // V: transposed bf16 output [HD][SEQ] per head (feeds PV contiguous staging)
    gemm_bf16_wmma<<<gqkv, 256, 0, stream>>>(
        xb, 0LL, Wvt, (long long)HD * DIM, bv, (long long)HD,
        nullptr, 0LL, nullptr, 0LL, Vtg, (long long)HD * SEQ,
        SEQ, HD, DIM, DIM, DIM, HD, SEQ);

    // flash attention -> concatenated heads (bf16)
    flash_attn_bf16<<<dim3(SEQ / 64, NH), 128, 0, stream>>>(Qb, Kb, Vtg, cat, scale);

    // output projection (f32 out)
    gemm_bf16_wmma<<<dim3(DIM / BN, SEQ / BM, 1), 256, 0, stream>>>(
        cat, 0LL, Wot, 0LL, bo, 0LL,
        attn_out, 0LL, nullptr, 0LL, nullptr, 0LL,
        SEQ, DIM, NH * HD, NH * HD, NH * HD, DIM, 0);

    // a = LN(x + attn_out)
    ln_residual<<<SEQ, 256, 0, stream>>>(x_f, attn_out, gamma, beta, a_f, ab);

    // ff = a @ Wf + bf
    gemm_bf16_wmma<<<dim3(DIM / BN, SEQ / BM, 1), 256, 0, stream>>>(
        ab, 0LL, Wft, 0LL, bff, 0LL,
        ff, 0LL, nullptr, 0LL, nullptr, 0LL,
        SEQ, DIM, DIM, DIM, DIM, DIM, 0);

    // x = LN(a + ff)
    ln_residual<<<SEQ, 256, 0, stream>>>(a_f, ff, gamma, beta, x_f, xb);
  }

  copy_f32<<<2048, 256, 0, stream>>>(x_f, out, SEQ * DIM);
}
